// Generator_20349555048890
// MI455X (gfx1250) — matrix operand order repaired and checked by reading, not compiled, not run on hardware
//
#include <hip/hip_runtime.h>
#include <hip/hip_bf16.h>

// ---------------------------------------------------------------------------
// Bi-LSTM greedy decoder for MI455X (gfx1250), wave32 + WMMA bf16 + TDM.
// VOCAB=10000 EMB=300(pad 320) HID=512 BS=64 PROMPT=8 MAXLEN=128
// ---------------------------------------------------------------------------

#define VOCAB   10000
#define EMB     300
#define EMBP    320
#define HID     512
#define BS      64
#define PROMPT  8
#define MAXLEN  128
#define NDEC    (MAXLEN - PROMPT)
#define G4      (4 * HID)            // 2048
#define NT_LOG  ((VOCAB + 15) / 16)  // 625 vocab column tiles

typedef __attribute__((ext_vector_type(16))) __bf16 v16bf;
typedef __attribute__((ext_vector_type(8)))  __bf16 v8bf;
typedef __attribute__((ext_vector_type(8)))  float  v8f;
typedef __attribute__((ext_vector_type(4)))  unsigned int u32x4;
typedef __attribute__((ext_vector_type(8)))  int i32x8;
typedef __attribute__((ext_vector_type(4)))  int i32x4;
typedef unsigned short u16;
typedef unsigned int   u32;
typedef unsigned long long u64;

// float -> bf16 bits, round to nearest even
__device__ __forceinline__ u16 f2bf(float f) {
  u32 u = __float_as_uint(f);
  u32 lsb = (u >> 16) & 1u;
  u += 0x7fffu + lsb;
  return (u16)(u >> 16);
}

// monotone float -> u32 (order-preserving for max)
__device__ __forceinline__ u32 ordf(float x) {
  u32 u = __float_as_uint(x);
  return (u & 0x80000000u) ? ~u : (u | 0x80000000u);
}

__device__ __forceinline__ float sigmoidf_(float x) {
  return 1.0f / (1.0f + __expf(-x));
}

// Build a 16-element bf16 fragment from two 128-bit loads.
// A-fragment (16x32, 16-bit): lane holds K=[k0,k0+8) and K=[k0+16,k0+24)  -> gap 16
// B-fragment (32x16, 16-bit): same K layout as A, one column              -> gap 16
__device__ __forceinline__ v16bf frag16(const u16* p, int gap) {
  v8bf lo = *(const v8bf*)p;
  v8bf hi = *(const v8bf*)(p + gap);
  v16bf r;
#pragma unroll
  for (int i = 0; i < 8; ++i) { r[i] = lo[i]; r[8 + i] = hi[i]; }
  return r;
}

__device__ __forceinline__ u64 max_shfl_xor_u64(u64 v, int m) {
  u32 lo = (u32)v, hi = (u32)(v >> 32);
  u32 olo = __shfl_xor(lo, m, 32);
  u32 ohi = __shfl_xor(hi, m, 32);
  u64 o = ((u64)ohi << 32) | olo;
  return o > v ? o : v;
}

// ---------------------------------------------------------------------------
// One-time init: convert weights to bf16, fuse biases, init h=1/c=0,
// clear argmax slots, copy prompt tokens into output.
// ---------------------------------------------------------------------------
__global__ void k_init(const int* __restrict__ tok,
                       const float* __restrict__ Wih, const float* __restrict__ Whh,
                       const float* __restrict__ bih, const float* __restrict__ bhh,
                       const float* __restrict__ fcW,
                       u16* __restrict__ fcW_bf, u16* __restrict__ Wih_bf,
                       u16* __restrict__ Whh_bf, float* __restrict__ bias,
                       u16* __restrict__ h0, float* __restrict__ c,
                       u64* __restrict__ amax, int* __restrict__ out) {
  const int tid = blockIdx.x * blockDim.x + threadIdx.x;
  const int stride = gridDim.x * blockDim.x;
  for (int i = tid; i < VOCAB * 2 * HID; i += stride)          // fc_W -> bf16
    fcW_bf[i] = f2bf(fcW[i]);
  for (int i = tid; i < 2 * G4 * HID; i += stride)             // W_hh -> bf16
    Whh_bf[i] = f2bf(Whh[i]);
  for (int i = tid; i < 2 * G4 * EMBP; i += stride) {          // W_ih -> bf16, pad K
    int r = i / EMBP, k = i - r * EMBP;
    Wih_bf[i] = (k < EMB) ? f2bf(Wih[r * EMB + k]) : (u16)0;
  }
  for (int i = tid; i < 2 * G4; i += stride)                   // fused bias
    bias[i] = bih[i] + bhh[i];
  for (int i = tid; i < 2 * BS * HID; i += stride) {           // h0=1 (bf16), c=0
    h0[i] = 0x3F80u;  // bf16(1.0)
    c[i]  = 0.0f;
  }
  for (int i = tid; i < BS; i += stride) amax[i] = 0ull;
  for (int i = tid; i < BS * PROMPT; i += stride) {            // emit prompt tokens
    int b = i >> 3, p = i & 7;
    out[b * MAXLEN + p] = tok[i];
  }
}

// ---------------------------------------------------------------------------
// Gather prompt-token embeddings into x_bf (bf16, K padded to 320)
// ---------------------------------------------------------------------------
__global__ void k_embed(const int* __restrict__ tok, const float* __restrict__ emb,
                        u16* __restrict__ x_bf, int t) {
  int i = blockIdx.x * blockDim.x + threadIdx.x;
  if (i >= BS * EMBP) return;
  int b = i / EMBP, k = i - b * EMBP;
  int tk = tok[b * PROMPT + t];
  x_bf[i] = (k < EMB) ? f2bf(emb[(size_t)tk * EMB + k]) : (u16)0;
}

// ---------------------------------------------------------------------------
// LSTM cell step: gates = x@Wihᵀ + h@Whhᵀ + b ; c,h update.
// 64 blocks x 4 waves = 256 units; unit = (dir, m_tile[0..3], j_tile[0..31]).
// Each wave computes the i/f/g/o 16x16 tiles at the SAME column slice, so the
// nonlinear cell update is wave-local. h double-buffered (bf16).
// ---------------------------------------------------------------------------
__global__ void __launch_bounds__(128) k_lstm(
    const u16* __restrict__ x_bf, const u16* __restrict__ h_in,
    u16* __restrict__ h_out, float* __restrict__ c,
    const u16* __restrict__ Wih_bf, const u16* __restrict__ Whh_bf,
    const float* __restrict__ bias) {
  const int lane = threadIdx.x & 31;
  const int u = blockIdx.x * 4 + (threadIdx.x >> 5);
  const int dir = u >> 7;
  const int rem = u & 127;
  const int mt = rem >> 5;        // M tile (batch rows)
  const int jt = rem & 31;        // column slice within HID

  v8f acc[4];
#pragma unroll
  for (int g = 0; g < 4; ++g) {
    float bv = bias[dir * G4 + g * HID + jt * 16 + (lane & 15)];
    acc[g] = (v8f){bv, bv, bv, bv, bv, bv, bv, bv};
  }

  const int am = mt * 16 + (lane & 15);        // A-matrix row for this lane
  const int ak = (lane & 16) ? 8 : 0;          // A K-offset per layout
  const int bn = lane & 15;                    // B column for this lane
  const int bk = (lane & 16) ? 8 : 0;          // B K-offset per layout

  const u16* WihD = Wih_bf + (size_t)dir * G4 * EMBP;
  const u16* WhhD = Whh_bf + (size_t)dir * G4 * HID;
  const u16* hinD = h_in + dir * BS * HID;

  // x @ W_ih^T  (K = 320)
#pragma unroll 2
  for (int kk = 0; kk < EMBP; kk += 32) {
    v16bf a = frag16(x_bf + am * EMBP + kk + ak, 16);
#pragma unroll
    for (int g = 0; g < 4; ++g) {
      v16bf b = frag16(WihD + (size_t)(g * HID + jt * 16 + bn) * EMBP + kk + bk, 16);
      acc[g] = __builtin_amdgcn_wmma_f32_16x16x32_bf16(
          false, a, false, b, (short)0, acc[g], false, false);
    }
  }
  // h @ W_hh^T  (K = 512)
#pragma unroll 2
  for (int kk = 0; kk < HID; kk += 32) {
    v16bf a = frag16(hinD + am * HID + kk + ak, 16);
#pragma unroll
    for (int g = 0; g < 4; ++g) {
      v16bf b = frag16(WhhD + (size_t)(g * HID + jt * 16 + bn) * HID + kk + bk, 16);
      acc[g] = __builtin_amdgcn_wmma_f32_16x16x32_bf16(
          false, a, false, b, (short)0, acc[g], false, false);
    }
  }

  // cell update: c' = sig(f)*c + sig(i)*tanh(g); h' = sig(o)*tanh(c')
  const int mrow0 = mt * 16 + ((lane & 16) ? 8 : 0);
  const int ncol  = jt * 16 + (lane & 15);
#pragma unroll
  for (int r = 0; r < 8; ++r) {
    int idx = dir * BS * HID + (mrow0 + r) * HID + ncol;
    float iv = sigmoidf_(acc[0][r]);
    float fv = sigmoidf_(acc[1][r]);
    float gv = tanhf(acc[2][r]);
    float ov = sigmoidf_(acc[3][r]);
    float cn = fv * c[idx] + iv * gv;
    c[idx] = cn;
    h_out[idx] = f2bf(ov * tanhf(cn));
  }
}

// ---------------------------------------------------------------------------
// logits = concat(h0,h1) @ fc_Wᵀ + fc_b, fused row-argmax.
// 157 blocks x 4 waves; wave owns one 16-wide vocab tile x all 4 M tiles.
// A-matrix (128 KB of h, bf16) is staged once per block into LDS by the
// Tensor Data Mover (tensor_load_to_lds + s_wait_tensorcnt): cuts per-step
// L2 A-traffic 4x and moves the 4-way-shared operand onto the LDS path while
// fc_W streams from L2 (resident: 20.5 MB bf16 of 192 MB).
// Argmax: pack (ordf(val)<<32)|~col, butterfly-max per 16-lane half,
// one atomicMax(u64) per batch row per wave. Ties -> lowest index (JAX).
// ---------------------------------------------------------------------------
__global__ void __launch_bounds__(128) k_logits(
    const u16* __restrict__ h_bf, const u16* __restrict__ fcW_bf,
    const float* __restrict__ fcb, u64* __restrict__ amax) {
  extern __shared__ u16 smem_a[];          // (2, 64, 512) bf16 = 128 KB
  const int lane = threadIdx.x & 31;
  const int nt = blockIdx.x * 4 + (threadIdx.x >> 5);

  // --- TDM stage: wave 0 issues one tensor DMA of the whole h image ---
  if (threadIdx.x < 32) {
    unsigned long long ga = (unsigned long long)(uintptr_t)h_bf;
    u32x4 g0;
    g0[0] = 1u;                                   // count=1 (valid user D#)
    g0[1] = 0u;                                   // lds_addr = 0
    g0[2] = (u32)(ga & 0xffffffffu);              // global_addr[31:0]
    g0[3] = (u32)((ga >> 32) & 0x01ffffffu) | (2u << 30);  // addr[56:32] | type=2
    i32x8 g1;
    g1[0] = (int)(1u << 16);        // wg_mask=0, data_size=1 (2 bytes/elem)
    g1[1] = (int)(512u << 16);      // tensor_dim0 = 512  (bits 79:48, low half)
    g1[2] = (int)(128u << 16);      // tensor_dim0 hi=0 | tensor_dim1 = 128
    g1[3] = (int)(512u << 16);      // tensor_dim1 hi=0 | tile_dim0 = 512
    g1[4] = 128;                    // tile_dim1 = 128, tile_dim2 = 0
    g1[5] = 512;                    // tensor_dim0_stride = 512 (lo)
    g1[6] = 0;                      // stride hi | dim1_stride lo
    g1[7] = 0;
    i32x4 z4 = (i32x4){0, 0, 0, 0};
    i32x8 z8 = (i32x8){0, 0, 0, 0, 0, 0, 0, 0};
    // 6-arg form (clang-23 / therock-10.0 toolchain)
    __builtin_amdgcn_tensor_load_to_lds(g0, g1, z4, z4, z8, 0);
    __builtin_amdgcn_s_wait_tensorcnt(0);
  }
  __syncthreads();
  if (nt >= NT_LOG) return;        // wave-uniform; after the barrier

  v8f acc[4];
#pragma unroll
  for (int m = 0; m < 4; ++m) acc[m] = (v8f){0, 0, 0, 0, 0, 0, 0, 0};

  const int ak = (lane & 16) ? 8 : 0;
  const int arow = lane & 15;
  const int bk = (lane & 16) ? 8 : 0;
  const u16* brow = fcW_bf + (size_t)(nt * 16 + (lane & 15)) * (2 * HID);

#pragma unroll 1
  for (int kk = 0; kk < 2 * HID; kk += 32) {
    __builtin_prefetch(brow + kk + 256, 0, 1);    // stream next chunk of fc_W row
    const int d = kk >> 9;                        // which direction half of concat
    const u16* abase = smem_a + d * BS * HID + (kk & (HID - 1));
    v16bf b = frag16(brow + kk + bk, 16);
#pragma unroll
    for (int m = 0; m < 4; ++m) {
      v16bf a = frag16(abase + (m * 16 + arow) * HID + ak, 16);  // ds_load_b128
      acc[m] = __builtin_amdgcn_wmma_f32_16x16x32_bf16(
          false, a, false, b, (short)0, acc[m], false, false);
    }
  }

  const int n = nt * 16 + (lane & 15);
  const float bn = (n < VOCAB) ? fcb[n] : 0.0f;
  const bool valid = (n < VOCAB);
#pragma unroll
  for (int m = 0; m < 4; ++m) {
#pragma unroll
    for (int r = 0; r < 8; ++r) {
      float v = acc[m][r] + bn;
      u64 key = valid ? ((((u64)ordf(v)) << 32) | (u32)(~(u32)n)) : 0ull;
      key = max_shfl_xor_u64(key, 1);
      key = max_shfl_xor_u64(key, 2);
      key = max_shfl_xor_u64(key, 4);
      key = max_shfl_xor_u64(key, 8);
      if ((lane & 15) == 0) {
        int row = m * 16 + r + ((lane & 16) ? 8 : 0);
        atomicMax(&amax[row], key);
      }
    }
  }
}

// ---------------------------------------------------------------------------
// Extract argmax token, emit it, gather its embedding into x_bf, reset slot.
// One block per batch row.
// ---------------------------------------------------------------------------
__global__ void k_finalize(const float* __restrict__ emb, u64* __restrict__ amax,
                           u16* __restrict__ x_bf, int* __restrict__ out, int t) {
  const int b = blockIdx.x;
  const u64 key = amax[b];                 // every thread reads before reset
  const int tok = (int)(~(u32)(key & 0xffffffffull));
  __syncthreads();
  if (threadIdx.x == 0) {
    out[b * MAXLEN + PROMPT + t] = tok;
    amax[b] = 0ull;
  }
  const float* er = emb + (size_t)tok * EMB;
  for (int k = threadIdx.x; k < EMBP; k += blockDim.x)
    x_bf[b * EMBP + k] = (k < EMB) ? f2bf(er[k]) : (u16)0;
}

// ---------------------------------------------------------------------------
extern "C" void kernel_launch(void* const* d_in, const int* in_sizes, int n_in,
                              void* d_out, int out_size, void* d_ws, size_t ws_size,
                              hipStream_t stream) {
  const int*   tok = (const int*)d_in[0];
  // d_in[1] = max_decode_len (constant 128, hardcoded)
  const float* emb = (const float*)d_in[2];
  const float* Wih = (const float*)d_in[3];
  const float* Whh = (const float*)d_in[4];
  const float* bih = (const float*)d_in[5];
  const float* bhh = (const float*)d_in[6];
  const float* fcW = (const float*)d_in[7];
  const float* fcb = (const float*)d_in[8];
  int* out = (int*)d_out;

  // workspace carve-up (256B aligned)
  char* w = (char*)d_ws;
  size_t off = 0;
  auto carve = [&](size_t bytes) -> void* {
    void* p = w + off;
    off = (off + bytes + 255) & ~(size_t)255;
    return p;
  };
  u16*   fcW_bf = (u16*)carve((size_t)VOCAB * 2 * HID * 2);    // 20.48 MB
  u16*   Wih_bf = (u16*)carve((size_t)2 * G4 * EMBP * 2);      //  2.62 MB
  u16*   Whh_bf = (u16*)carve((size_t)2 * G4 * HID * 2);       //  4.19 MB
  float* bias   = (float*)carve((size_t)2 * G4 * 4);
  u16*   hbuf0  = (u16*)carve((size_t)2 * BS * HID * 2);
  u16*   hbuf1  = (u16*)carve((size_t)2 * BS * HID * 2);
  float* cst    = (float*)carve((size_t)2 * BS * HID * 4);
  u16*   x_bf   = (u16*)carve((size_t)BS * EMBP * 2);
  u64*   amax   = (u64*)carve((size_t)BS * 8);
  u16*   hbuf[2] = {hbuf0, hbuf1};

  k_init<<<1024, 256, 0, stream>>>(tok, Wih, Whh, bih, bhh, fcW,
                                   fcW_bf, Wih_bf, Whh_bf, bias,
                                   hbuf0, cst, amax, out);

  int pb = 0;
  // prompt phase: advance hidden state through given tokens
  for (int t = 0; t < PROMPT; ++t) {
    k_embed<<<(BS * EMBP + 255) / 256, 256, 0, stream>>>(tok, emb, x_bf, t);
    k_lstm<<<64, 128, 0, stream>>>(x_bf, hbuf[pb], hbuf[pb ^ 1], cst,
                                   Wih_bf, Whh_bf, bias);
    pb ^= 1;
  }
  // greedy decode phase
  const size_t lds_bytes = (size_t)2 * BS * HID * sizeof(u16);  // 128 KB
  for (int t = 0; t < NDEC; ++t) {
    k_lstm<<<64, 128, 0, stream>>>(x_bf, hbuf[pb], hbuf[pb ^ 1], cst,
                                   Wih_bf, Whh_bf, bias);
    pb ^= 1;
    k_logits<<<(NT_LOG + 3) / 4, 128, lds_bytes, stream>>>(hbuf[pb], fcW_bf,
                                                           fcb, amax);
    k_finalize<<<BS, 256, 0, stream>>>(emb, amax, x_bf, out, t);
  }
}
